// RoPEAttention_51891794870893
// MI455X (gfx1250) — matrix operand order repaired and checked by reading, not compiled, not run on hardware
//
#include <hip/hip_runtime.h>
#include <hip/hip_bf16.h>

#define Bb 4
#define Nn 2048
#define Cc 1024
#define Hh 16
#define Dd 64
#define MM (Bb*Nn)      // 8192 flattened rows
#define K3C (3*Cc)      // 3072 qkv output channels

typedef __attribute__((ext_vector_type(16))) __bf16 v16bf;
typedef __attribute__((ext_vector_type(8)))  __bf16 v8bf;
typedef __attribute__((ext_vector_type(8)))  float  v8f;

__device__ __forceinline__ v8f vzero() {
    v8f z;
#pragma unroll
    for (int i = 0; i < 8; ++i) z[i] = 0.f;
    return z;
}

__device__ __forceinline__ v8f wmma_bf16(v16bf a, v16bf b, v8f c) {
    // emits v_wmma_f32_16x16x32_bf16
    return __builtin_amdgcn_wmma_f32_16x16x32_bf16(
        false, a, false, b, (short)0, c, false, false);
}

// A-matrix (16-bit, 16x32, MxK) per-lane fragment: lane holds row M=lane&15;
// elements 0..7 at K = k0 + 16*half + e, elements 8..15 at K = k0 + 16*half + 8 + e.
__device__ __forceinline__ v16bf load_a_frag(const __bf16* row, int k0, int half) {
    const v8bf lo = *(const v8bf*)(row + k0 + 16 * half);
    const v8bf hi = *(const v8bf*)(row + k0 + 16 * half + 8);
    v16bf a;
#pragma unroll
    for (int i = 0; i < 8; ++i) { a[i] = lo[i]; a[i + 8] = hi[i]; }
    return a;
}

// ---------------------------------------------------------------------------
// f32 -> bf16 conversion (grid-stride)
// ---------------------------------------------------------------------------
__global__ void cvt_f32_bf16(const float* __restrict__ in, __bf16* __restrict__ out, int n) {
    int i = blockIdx.x * blockDim.x + threadIdx.x;
    int stride = gridDim.x * blockDim.x;
    for (; i < n; i += stride) out[i] = (__bf16)in[i];
}

// ---------------------------------------------------------------------------
// QKV GEMM, 64x64 tile per wave (4 M-subtiles x 4 N-subtiles, K=1024),
// fused with per-head LayerNorm + RoPE. grid = (MM/64, 48):
// blockIdx.y = which*16 + h; the 64-wide N tile covers one head's D=64.
// Writes q,k as bf16 [b,h,n,d]; v transposed as bf16 [b,h,d,n] (q pre-scaled).
// ---------------------------------------------------------------------------
__global__ void __launch_bounds__(32)
qkv_gemm_lnrope_kernel(const __bf16* __restrict__ xb, const __bf16* __restrict__ wb,
                       const float* __restrict__ qg, const float* __restrict__ qbeta,
                       const float* __restrict__ kg, const float* __restrict__ kbeta,
                       __bf16* __restrict__ qo, __bf16* __restrict__ ko,
                       __bf16* __restrict__ vto) {
    const int lane = threadIdx.x;
    const int col  = lane & 15;
    const int half = lane >> 4;
    const int m0   = blockIdx.x * 64;
    const int g    = blockIdx.y;     // 0..47
    const int which = g >> 4;        // 0=q, 1=k, 2=v
    const int h     = g & 15;
    const int obase = g * 64;        // == which*C + h*D

    const __bf16* arow[4];
#pragma unroll
    for (int mi = 0; mi < 4; ++mi)
        arow[mi] = xb + (size_t)(m0 + 16 * mi + col) * Cc;

    v8f acc[4][4];
#pragma unroll
    for (int mi = 0; mi < 4; ++mi)
#pragma unroll
        for (int t = 0; t < 4; ++t) acc[mi][t] = vzero();

    for (int k0 = 0; k0 < Cc; k0 += 32) {
        v16bf a[4];
#pragma unroll
        for (int mi = 0; mi < 4; ++mi) a[mi] = load_a_frag(arow[mi], k0, half);
        v16bf bfr[4];
#pragma unroll
        for (int t = 0; t < 4; ++t)
            bfr[t] = *(const v16bf*)(wb + (size_t)(obase + 16 * t + col) * Cc + k0 + 16 * half);
#pragma unroll
        for (int mi = 0; mi < 4; ++mi)
#pragma unroll
            for (int t = 0; t < 4; ++t)
                acc[mi][t] = wmma_bf16(a[mi], bfr[t], acc[mi][t]);
    }

    // ----- epilogue: per M-subtile LayerNorm over D=64, RoPE, scale, store ---
    float gam[4], bet[4];
    if (which < 2) {
        const float* gsrc = (which == 0) ? qg : kg;
        const float* bsrc = (which == 0) ? qbeta : kbeta;
#pragma unroll
        for (int t = 0; t < 4; ++t) { gam[t] = gsrc[16 * t + col]; bet[t] = bsrc[16 * t + col]; }
    }
    const float LOGB = 9.210340371976184f;  // ln(10000)
    const float f0 = __expf(-((float)col)        * (LOGB / 32.f));
    const float f1 = __expf(-((float)(col + 16)) * (LOGB / 32.f));

#pragma unroll
    for (int mi = 0; mi < 4; ++mi) {
        const int mrow0 = m0 + 16 * mi;
        if (which < 2) {
#pragma unroll
            for (int v = 0; v < 8; ++v) {
                float s  = acc[mi][0][v] + acc[mi][1][v] + acc[mi][2][v] + acc[mi][3][v];
                float ss = acc[mi][0][v] * acc[mi][0][v] + acc[mi][1][v] * acc[mi][1][v]
                         + acc[mi][2][v] * acc[mi][2][v] + acc[mi][3][v] * acc[mi][3][v];
#pragma unroll
                for (int msk = 1; msk < 16; msk <<= 1) {
                    s  += __shfl_xor(s,  msk, 16);
                    ss += __shfl_xor(ss, msk, 16);
                }
                float mu  = s * (1.f / 64.f);
                float var = ss * (1.f / 64.f) - mu * mu;
                float inv = rsqrtf(var + 1e-6f);
#pragma unroll
                for (int t = 0; t < 4; ++t)
                    acc[mi][t][v] = (acc[mi][t][v] - mu) * inv * gam[t] + bet[t];
            }
            // RoPE: freq index j = d & 31; pairs d <-> d+32 live in tiles t <-> t^2.
#pragma unroll
            for (int v = 0; v < 8; ++v) {
                float n = (float)(mrow0 + v + 8 * half);
                float sn0, cs0, sn1, cs1;
                __sincosf(n * f0, &sn0, &cs0);
                __sincosf(n * f1, &sn1, &cs1);
                float x0 = acc[mi][0][v], x1 = acc[mi][1][v];
                float x2 = acc[mi][2][v], x3 = acc[mi][3][v];
                acc[mi][0][v] = x0 * cs0 - x2 * sn0;   // d < 32  : x*cos - x[d+32]*sin
                acc[mi][1][v] = x1 * cs1 - x3 * sn1;
                acc[mi][2][v] = x2 * cs0 + x0 * sn0;   // d >= 32 : x*cos + x[d-32]*sin
                acc[mi][3][v] = x3 * cs1 + x1 * sn1;
            }
            if (which == 0) {
#pragma unroll
                for (int t = 0; t < 4; ++t)
#pragma unroll
                    for (int v = 0; v < 8; ++v) acc[mi][t][v] *= 0.125f;  // D^-0.5
            }
        }

        const int bidx = mrow0 >> 11;      // / 2048 (tiles never cross batch rows)
        const int nloc = mrow0 & 2047;
        if (which == 2) {
            // v transposed: [b,h,d,n]; lane's 8 rows are consecutive n -> 16B store
#pragma unroll
            for (int t = 0; t < 4; ++t) {
                int d = 16 * t + col;
                v8bf pack;
#pragma unroll
                for (int v = 0; v < 8; ++v) pack[v] = (__bf16)acc[mi][t][v];
                *(v8bf*)(vto + ((size_t)((bidx * Hh + h) * Dd + d)) * Nn + nloc + 8 * half) = pack;
            }
        } else {
            __bf16* base = (which == 0) ? qo : ko;
#pragma unroll
            for (int v = 0; v < 8; ++v) {
                size_t off = ((size_t)((bidx * Hh + h) * Nn) + nloc + v + 8 * half) * Dd;
#pragma unroll
                for (int t = 0; t < 4; ++t) base[off + 16 * t + col] = (__bf16)acc[mi][t][v];
            }
        }
    }
}

// ---------------------------------------------------------------------------
// Flash attention: one wave per (b, h, 32 q-rows); online softmax over
// key blocks of 32. K/V fragments are shared across both q-row subtiles.
// P transposed C-layout -> A-layout through LDS (in-order DS ops within a
// wave, no barrier needed). Output bf16 [b,n,h,d] = [b,n,C].
// ---------------------------------------------------------------------------
__global__ void __launch_bounds__(32)
attn_kernel(const __bf16* __restrict__ q, const __bf16* __restrict__ k,
            const __bf16* __restrict__ vt, __bf16* __restrict__ out) {
    const int lane = threadIdx.x;
    const int col  = lane & 15;
    const int half = lane >> 4;
    const int n0 = blockIdx.x * 32;
    const int h  = blockIdx.y;
    const int b  = blockIdx.z;

    const __bf16* qh = q  + ((size_t)(b * Hh + h)) * Nn * Dd;
    const __bf16* kh = k  + ((size_t)(b * Hh + h)) * Nn * Dd;
    const __bf16* vh = vt + ((size_t)(b * Hh + h)) * Dd * Nn;

    v16bf aq[2][2];
#pragma unroll
    for (int mi = 0; mi < 2; ++mi) {
        aq[mi][0] = load_a_frag(qh + (size_t)(n0 + 16 * mi + col) * Dd, 0,  half);
        aq[mi][1] = load_a_frag(qh + (size_t)(n0 + 16 * mi + col) * Dd, 32, half);
    }

    v8f acc[2][4];
    float mrow[2][8], lrow[2][8];
#pragma unroll
    for (int mi = 0; mi < 2; ++mi) {
#pragma unroll
        for (int t = 0; t < 4; ++t) acc[mi][t] = vzero();
#pragma unroll
        for (int v = 0; v < 8; ++v) { mrow[mi][v] = -3.0e38f; lrow[mi][v] = 0.f; }
    }

    __shared__ __align__(32) __bf16 plds[2][16][32];

    for (int j0 = 0; j0 < Nn; j0 += 32) {
        // S = q . k^T : 32 rows x 32 keys (K frags shared by both M-subtiles)
        v8f s[2][2];
#pragma unroll
        for (int mi = 0; mi < 2; ++mi) { s[mi][0] = vzero(); s[mi][1] = vzero(); }
        {
            v16bf b0 = *(const v16bf*)(kh + (size_t)(j0 + col) * Dd + 16 * half);
            v16bf b1 = *(const v16bf*)(kh + (size_t)(j0 + 16 + col) * Dd + 16 * half);
#pragma unroll
            for (int mi = 0; mi < 2; ++mi) {
                s[mi][0] = wmma_bf16(aq[mi][0], b0, s[mi][0]);
                s[mi][1] = wmma_bf16(aq[mi][0], b1, s[mi][1]);
            }
            v16bf c0 = *(const v16bf*)(kh + (size_t)(j0 + col) * Dd + 32 + 16 * half);
            v16bf c1 = *(const v16bf*)(kh + (size_t)(j0 + 16 + col) * Dd + 32 + 16 * half);
#pragma unroll
            for (int mi = 0; mi < 2; ++mi) {
                s[mi][0] = wmma_bf16(aq[mi][1], c0, s[mi][0]);
                s[mi][1] = wmma_bf16(aq[mi][1], c1, s[mi][1]);
            }
        }
        // online softmax per row (row = v + 8*half; 16-lane xor tree per half)
#pragma unroll
        for (int mi = 0; mi < 2; ++mi) {
#pragma unroll
            for (int v = 0; v < 8; ++v) {
                float bm = fmaxf(s[mi][0][v], s[mi][1][v]);
#pragma unroll
                for (int msk = 1; msk < 16; msk <<= 1)
                    bm = fmaxf(bm, __shfl_xor(bm, msk, 16));
                float mn = fmaxf(mrow[mi][v], bm);
                float alpha = __expf(mrow[mi][v] - mn);
                float p0 = __expf(s[mi][0][v] - mn);
                float p1 = __expf(s[mi][1][v] - mn);
                float rs = p0 + p1;
#pragma unroll
                for (int msk = 1; msk < 16; msk <<= 1)
                    rs += __shfl_xor(rs, msk, 16);
                lrow[mi][v] = lrow[mi][v] * alpha + rs;
                mrow[mi][v] = mn;
#pragma unroll
                for (int t = 0; t < 4; ++t) acc[mi][t][v] *= alpha;
                int r = v + 8 * half;
                plds[mi][r][col]      = (__bf16)p0;
                plds[mi][r][16 + col] = (__bf16)p1;
            }
        }
        // P: C-layout -> A-layout (16x32) via LDS, per M-subtile
        v16bf ap[2];
#pragma unroll
        for (int mi = 0; mi < 2; ++mi) {
            const v8bf lo = *(const v8bf*)(&plds[mi][col][16 * half]);
            const v8bf hi = *(const v8bf*)(&plds[mi][col][16 * half + 8]);
#pragma unroll
            for (int i = 0; i < 8; ++i) { ap[mi][i] = lo[i]; ap[mi][i + 8] = hi[i]; }
        }
        // acc += P @ V  (V frag loaded once, used by both M-subtiles)
#pragma unroll
        for (int t = 0; t < 4; ++t) {
            v16bf bv = *(const v16bf*)(vh + (size_t)(16 * t + col) * Nn + j0 + 16 * half);
#pragma unroll
            for (int mi = 0; mi < 2; ++mi)
                acc[mi][t] = wmma_bf16(ap[mi], bv, acc[mi][t]);
        }
    }

#pragma unroll
    for (int mi = 0; mi < 2; ++mi) {
#pragma unroll
        for (int v = 0; v < 8; ++v) {
            float invl = 1.f / lrow[mi][v];
            int r = v + 8 * half;
            size_t off = ((size_t)b * Nn + n0 + 16 * mi + r) * Cc + h * Dd;
#pragma unroll
            for (int t = 0; t < 4; ++t)
                out[off + 16 * t + col] = (__bf16)(acc[mi][t][v] * invl);
        }
    }
}

// ---------------------------------------------------------------------------
// Output projection: 64x64 tile per wave, K=1024, + bias, f32 result.
// ---------------------------------------------------------------------------
__global__ void __launch_bounds__(32)
proj_gemm_kernel(const __bf16* __restrict__ a, const __bf16* __restrict__ w,
                 const float* __restrict__ bias, float* __restrict__ out) {
    const int lane = threadIdx.x;
    const int col  = lane & 15;
    const int half = lane >> 4;
    const int m0    = blockIdx.x * 64;
    const int obase = blockIdx.y * 64;

    const __bf16* arow[4];
#pragma unroll
    for (int mi = 0; mi < 4; ++mi)
        arow[mi] = a + (size_t)(m0 + 16 * mi + col) * Cc;

    v8f acc[4][4];
#pragma unroll
    for (int mi = 0; mi < 4; ++mi)
#pragma unroll
        for (int t = 0; t < 4; ++t) acc[mi][t] = vzero();

    for (int k0 = 0; k0 < Cc; k0 += 32) {
        v16bf av[4];
#pragma unroll
        for (int mi = 0; mi < 4; ++mi) av[mi] = load_a_frag(arow[mi], k0, half);
        v16bf bfr[4];
#pragma unroll
        for (int t = 0; t < 4; ++t)
            bfr[t] = *(const v16bf*)(w + (size_t)(obase + 16 * t + col) * Cc + k0 + 16 * half);
#pragma unroll
        for (int mi = 0; mi < 4; ++mi)
#pragma unroll
            for (int t = 0; t < 4; ++t)
                acc[mi][t] = wmma_bf16(av[mi], bfr[t], acc[mi][t]);
    }

    float bs[4];
#pragma unroll
    for (int t = 0; t < 4; ++t) bs[t] = bias[obase + 16 * t + col];
#pragma unroll
    for (int mi = 0; mi < 4; ++mi) {
#pragma unroll
        for (int v = 0; v < 8; ++v) {
            size_t row = (size_t)m0 + 16 * mi + v + 8 * half;
#pragma unroll
            for (int t = 0; t < 4; ++t)
                out[row * Cc + obase + 16 * t + col] = acc[mi][t][v] + bs[t];
        }
    }
}

// ---------------------------------------------------------------------------
extern "C" void kernel_launch(void* const* d_in, const int* in_sizes, int n_in,
                              void* d_out, int out_size, void* d_ws, size_t ws_size,
                              hipStream_t stream) {
    (void)in_sizes; (void)n_in; (void)out_size; (void)ws_size;
    const float* x      = (const float*)d_in[0];
    const float* qkv_w  = (const float*)d_in[1];
    const float* qg     = (const float*)d_in[2];
    const float* qb     = (const float*)d_in[3];
    const float* kg     = (const float*)d_in[4];
    const float* kb     = (const float*)d_in[5];
    const float* pw     = (const float*)d_in[6];
    const float* pb     = (const float*)d_in[7];
    float* out          = (float*)d_out;

    char* ws = (char*)d_ws;
    size_t off = 0;
    auto carve = [&](size_t elems) {
        void* p = ws + off;
        off += (elems * sizeof(__bf16) + 255) & ~(size_t)255;
        return p;
    };
    __bf16* xb    = (__bf16*)carve((size_t)MM * Cc);    // x in bf16
    __bf16* wqkvb = (__bf16*)carve((size_t)K3C * Cc);   // qkv_w in bf16
    __bf16* pwb   = (__bf16*)carve((size_t)Cc * Cc);    // proj_w in bf16
    __bf16* qo    = (__bf16*)carve((size_t)MM * Cc);    // q  [b,h,n,d]
    __bf16* ko    = (__bf16*)carve((size_t)MM * Cc);    // k  [b,h,n,d]
    __bf16* vto   = (__bf16*)carve((size_t)MM * Cc);    // vT [b,h,d,n]
    __bf16* attnb = (__bf16*)carve((size_t)MM * Cc);    // attention out [b,n,C]

    cvt_f32_bf16<<<4096, 256, 0, stream>>>(x, xb, MM * Cc);
    cvt_f32_bf16<<<2048, 256, 0, stream>>>(qkv_w, wqkvb, K3C * Cc);
    cvt_f32_bf16<<<1024, 256, 0, stream>>>(pw, pwb, Cc * Cc);

    dim3 g1(MM / 64, 48);
    qkv_gemm_lnrope_kernel<<<g1, 32, 0, stream>>>(xb, wqkvb, qg, qb, kg, kb, qo, ko, vto);

    dim3 g2(Nn / 32, Hh, Bb);
    attn_kernel<<<g2, 32, 0, stream>>>(qo, ko, vto, attnb);

    dim3 g3(MM / 64, Cc / 64);
    proj_gemm_kernel<<<g3, 32, 0, stream>>>(attnb, pwb, pb, out);
}
